// MeshCNNClassifier_17386027614271
// MI455X (gfx1250) — matrix operand order repaired and checked by reading, not compiled, not run on hardware
//
#include <hip/hip_runtime.h>

typedef unsigned short u16;
typedef unsigned int   u32;

typedef __attribute__((ext_vector_type(16))) __bf16 v16bf;
typedef __attribute__((ext_vector_type(8)))  float  v8f;

__device__ __forceinline__ float bf2f(u32 u) {
  union { u32 i; float f; } c; c.i = u << 16; return c.f;
}
__device__ __forceinline__ u16 f2bf(float f) {
  union { float f; u32 i; } c; c.f = f;
  u32 r = c.i + 0x7FFFu + ((c.i >> 16) & 1u);
  return (u16)(r >> 16);
}

// elementwise min/max of two packed-bf16 dwords (matches jnp.sort of a pair)
__device__ __forceinline__ void bfmm(u32 a, u32 b, u32& mn, u32& mx) {
  float al = bf2f(a & 0xFFFFu), ah = bf2f(a >> 16);
  float bl = bf2f(b & 0xFFFFu), bh = bf2f(b >> 16);
  u32 mnl = (al <= bl) ? (a & 0xFFFFu) : (b & 0xFFFFu);
  u32 mxl = (al <= bl) ? (b & 0xFFFFu) : (a & 0xFFFFu);
  u32 mnh = (ah <= bh) ? (a >> 16) : (b >> 16);
  u32 mxh = (ah <= bh) ? (b >> 16) : (a >> 16);
  mn = mnl | (mnh << 16);
  mx = mxl | (mxh << 16);
}
__device__ __forceinline__ void bfmm4(uint4 a, uint4 b, uint4& mn, uint4& mx) {
  bfmm(a.x, b.x, mn.x, mx.x); bfmm(a.y, b.y, mn.y, mx.y);
  bfmm(a.z, b.z, mn.z, mx.z); bfmm(a.w, b.w, mn.w, mx.w);
}

// ---------------- weight fp32 -> bf16 prep ----------------
__global__ void prep_wbf16(const float* __restrict__ src, u16* __restrict__ dst, int n) {
  int i = blockIdx.x * 256 + threadIdx.x;
  if (i < n) dst[i] = f2bf(src[i]);
}

// ---------------- layer 0: K = 55 (fp32 VALU, weights scalar-cached) ----------------
__global__ __launch_bounds__(256) void layer0_kernel(
    const float* __restrict__ x, const int* __restrict__ nbr,
    const float* __restrict__ W0, const float* __restrict__ b0,
    const float* __restrict__ gamma, const float* __restrict__ beta,
    u16* __restrict__ curOut, int E)
{
  __shared__ float sH[256 * 65];   // padded per-thread rows (bank-conflict free)
  long e  = (long)blockIdx.x * 256 + threadIdx.x;
  bool ok = e < (long)E;
  long ec = ok ? e : (long)E - 1;

  float comb[55];
#pragma unroll
  for (int c = 0; c < 11; ++c) comb[c] = x[ec * 11 + c];

  int4 ni = *(const int4*)(&nbr[ec * 4]);
  int ids[4] = { ni.x, ni.y, ni.z, ni.w };
  float nf[4][11];
#pragma unroll
  for (int j = 0; j < 4; ++j) {
    long nj = ids[j];
#pragma unroll
    for (int c = 0; c < 11; ++c) nf[j][c] = (ids[j] < 0) ? 0.f : x[nj * 11 + c];
  }
#pragma unroll
  for (int c = 0; c < 11; ++c) {
    comb[11 + c] = fminf(nf[0][c], nf[1][c]);
    comb[22 + c] = fmaxf(nf[0][c], nf[1][c]);
    comb[33 + c] = fminf(nf[2][c], nf[3][c]);
    comb[44 + c] = fmaxf(nf[2][c], nf[3][c]);
  }

  float* hrow = &sH[threadIdx.x * 65];
  float s = 0.f, ss = 0.f;
  for (int n = 0; n < 64; ++n) {
    float a = b0[n];
#pragma unroll
    for (int k = 0; k < 55; ++k) a += comb[k] * W0[n * 55 + k];   // uniform -> SMEM
    hrow[n] = a; s += a; ss += a * a;
  }
  float mu  = s * (1.f / 64.f);
  float var = ss * (1.f / 64.f) - mu * mu;
  float inv = rsqrtf(var + 1e-5f);

  if (ok) {
    u16 outv[64];
#pragma unroll
    for (int n = 0; n < 64; ++n) {
      float y = (hrow[n] - mu) * inv * gamma[n] + beta[n];
      outv[n] = f2bf(fmaxf(y, 0.f));
    }
#pragma unroll
    for (int q = 0; q < 8; ++q)
      *(uint4*)(curOut + e * 64 + q * 8) = ((const uint4*)outv)[q];
  }
}

// ---------------- layers 1..3: gathered GEMM via bf16 WMMA ----------------
// 128 edges/workgroup, 8 waves x (16-edge M-tile, N=64, K=320 in 10 steps of 32)
#define AST 328   // LDS A / B row stride in bf16 (164 dwords, 36 mod 64 -> conflict free)
#define CST 67    // LDS C row stride in f32

union Frag { uint4 q[2]; v16bf v; };

__device__ __forceinline__ void load_frags(const u16* pa, const u16* pb, int kk, int hh,
                                           Frag& A, Frag& B0, Frag& B1, Frag& B2, Frag& B3) {
  A.q[0] = *(const uint4*)(pa + kk * 32 + hh * 16);       // e0..7  -> k = kk*32 + h*16 + e
  A.q[1] = *(const uint4*)(pa + kk * 32 + hh * 16 + 8);   // e8..15 -> k = kk*32 + h*16 + e
  const u16* p0 = pb + kk * 32;                           // B: k = kk*32 + h*16 + e (contiguous)
  B0.q[0] = *(const uint4*)(p0);
  B0.q[1] = *(const uint4*)(p0 + 8);
  B1.q[0] = *(const uint4*)(p0 + 16 * AST);
  B1.q[1] = *(const uint4*)(p0 + 16 * AST + 8);
  B2.q[0] = *(const uint4*)(p0 + 32 * AST);
  B2.q[1] = *(const uint4*)(p0 + 32 * AST + 8);
  B3.q[0] = *(const uint4*)(p0 + 48 * AST);
  B3.q[1] = *(const uint4*)(p0 + 48 * AST + 8);
}

__global__ __launch_bounds__(256) void mesh_layer_kernel(
    const u16* __restrict__ curIn, u16* __restrict__ curOut,
    const int* __restrict__ nbr, const u16* __restrict__ Wb,
    const float* __restrict__ bias, const float* __restrict__ gamma,
    const float* __restrict__ beta, int E)
{
  __shared__ __align__(16) u16 sA[128 * AST];   // combined features (reused as f32 C)
  __shared__ __align__(16) u16 sB[64 * AST];    // W bf16, row-major [64][320]

  const int t = threadIdx.x;
  const long tileBase = (long)blockIdx.x * 128;

  // stage weights (uint4 granularity): 64*320/8 = 2560 chunks
  {
    const uint4* src = (const uint4*)Wb;
    for (int i = t; i < 2560; i += 256) {
      int n = i / 40, kb = i % 40;
      *(uint4*)(&sB[n * AST + kb * 8]) = src[i];
    }
  }

  // build combined rows: thread t -> edge t/2, channel half (t&1)*32
  {
    int  eL = t >> 1;
    int  cb = (t & 1) * 32;
    long e  = tileBase + eL;
    if (e >= (long)E) e = (long)E - 1;

    const uint4* sp = (const uint4*)(curIn + e * 64 + cb);
    uint4 selfq[4] = { sp[0], sp[1], sp[2], sp[3] };

    int4 ni = *(const int4*)(&nbr[e * 4]);
    int ids[4] = { ni.x, ni.y, ni.z, ni.w };
    uint4 nfq[4][4];
#pragma unroll
    for (int j = 0; j < 4; ++j) {
      if (ids[j] < 0) {
        nfq[j][0] = nfq[j][1] = nfq[j][2] = nfq[j][3] = make_uint4(0, 0, 0, 0);
      } else {
        const uint4* p = (const uint4*)(curIn + (long)ids[j] * 64 + cb);
        nfq[j][0] = p[0]; nfq[j][1] = p[1]; nfq[j][2] = p[2]; nfq[j][3] = p[3];
      }
    }
    u16* row = &sA[eL * AST];
#pragma unroll
    for (int q = 0; q < 4; ++q) {
      uint4 mn01, mx01, mn23, mx23;
      bfmm4(nfq[0][q], nfq[1][q], mn01, mx01);
      bfmm4(nfq[2][q], nfq[3][q], mn23, mx23);
      *(uint4*)(row +       cb + q * 8) = selfq[q];
      *(uint4*)(row +  64 + cb + q * 8) = mn01;
      *(uint4*)(row + 128 + cb + q * 8) = mx01;
      *(uint4*)(row + 192 + cb + q * 8) = mn23;
      *(uint4*)(row + 256 + cb + q * 8) = mx23;
    }
  }
  __syncthreads();

  // WMMA: wave w -> edges [w*16, w*16+16), all 64 output channels.
  // Software-pipelined: prefetch next k-step's fragments before current WMMAs.
  const int wave = t >> 5, lane = t & 31;
  const int hh = lane >> 4, mm = lane & 15;
  const u16* pa = &sA[(wave * 16 + mm) * AST];
  const u16* pb = &sB[mm * AST + hh * 16];

  v8f acc0 = {}, acc1 = {}, acc2 = {}, acc3 = {};
  Frag A, B0, B1, B2, B3, An, B0n, B1n, B2n, B3n;
  load_frags(pa, pb, 0, hh, A, B0, B1, B2, B3);
#pragma unroll
  for (int kk = 0; kk < 10; ++kk) {
    if (kk < 9) load_frags(pa, pb, kk + 1, hh, An, B0n, B1n, B2n, B3n);
    acc0 = __builtin_amdgcn_wmma_f32_16x16x32_bf16(false, A.v, false, B0.v, (short)0, acc0, false, false);
    acc1 = __builtin_amdgcn_wmma_f32_16x16x32_bf16(false, A.v, false, B1.v, (short)0, acc1, false, false);
    acc2 = __builtin_amdgcn_wmma_f32_16x16x32_bf16(false, A.v, false, B2.v, (short)0, acc2, false, false);
    acc3 = __builtin_amdgcn_wmma_f32_16x16x32_bf16(false, A.v, false, B3.v, (short)0, acc3, false, false);
    A = An; B0 = B0n; B1 = B1n; B2 = B2n; B3 = B3n;   // renamed away by full unroll
  }
  __syncthreads();   // all waves done reading sA before it becomes sC

  // scatter accumulators: lane holds n = nt*16 + mm, VGPR r -> m = r + 8*hh
  float* sC = (float*)sA;
#pragma unroll
  for (int r = 0; r < 8; ++r) {
    int m = wave * 16 + r + 8 * hh;
    sC[m * CST + 0  + mm] = acc0[r];
    sC[m * CST + 16 + mm] = acc1[r];
    sC[m * CST + 32 + mm] = acc2[r];
    sC[m * CST + 48 + mm] = acc3[r];
  }
  __syncthreads();

  // bias + LayerNorm + ReLU + residual, one edge per thread (threads 0..127)
  if (t < 128) {
    long e = tileBase + t;
    if (e < (long)E) {
      const float* row = (const float*)sA + t * CST;
      float v[64];
      float s = 0.f, ss = 0.f;
#pragma unroll
      for (int n = 0; n < 64; ++n) {
        float y = row[n] + bias[n];
        v[n] = y; s += y; ss += y * y;
      }
      float mu  = s * (1.f / 64.f);
      float var = ss * (1.f / 64.f) - mu * mu;
      float inv = rsqrtf(var + 1e-5f);
      const u16* selfr = curIn + e * 64;
      u16 outv[64];
#pragma unroll
      for (int n = 0; n < 64; ++n) {
        float y = (v[n] - mu) * inv * gamma[n] + beta[n];
        y = fmaxf(y, 0.f) + bf2f(selfr[n]);      // residual
        outv[n] = f2bf(y);
      }
#pragma unroll
      for (int q = 0; q < 8; ++q)
        *(uint4*)(curOut + e * 64 + q * 8) = ((const uint4*)outv)[q];
    }
  }
}

// ---------------- classifier head: 64 -> 32 -> 1 ----------------
__global__ __launch_bounds__(256) void classifier_kernel(
    const u16* __restrict__ cur, const float* __restrict__ Wc1,
    const float* __restrict__ bc1, const float* __restrict__ Wc2,
    const float* __restrict__ bc2, float* __restrict__ out, int E)
{
  long e = (long)blockIdx.x * 256 + threadIdx.x;
  if (e >= (long)E) return;
  const u16* r = cur + e * 64;
  float v[64];
#pragma unroll
  for (int n = 0; n < 64; ++n) v[n] = bf2f(r[n]);
  float o = bc2[0];
  for (int j = 0; j < 32; ++j) {
    float a = bc1[j];
#pragma unroll
    for (int k = 0; k < 64; ++k) a += v[k] * Wc1[j * 64 + k];   // uniform -> SMEM
    o += fmaxf(a, 0.f) * Wc2[j];
  }
  out[e] = o;
}

extern "C" void kernel_launch(void* const* d_in, const int* in_sizes, int n_in,
                              void* d_out, int out_size, void* d_ws, size_t ws_size,
                              hipStream_t stream) {
  const float* x      = (const float*)d_in[0];
  const int*   nbr    = (const int*)d_in[1];      // int32 (JAX x64 disabled)
  const float* W0     = (const float*)d_in[2];
  const float* b0     = (const float*)d_in[3];
  const float* W_rest = (const float*)d_in[4];    // [3][64][320]
  const float* b_rest = (const float*)d_in[5];    // [3][64]
  const float* gamma  = (const float*)d_in[6];    // [4][64]
  const float* beta   = (const float*)d_in[7];    // [4][64]
  const float* Wc1    = (const float*)d_in[8];
  const float* bc1    = (const float*)d_in[9];
  const float* Wc2    = (const float*)d_in[10];
  const float* bc2    = (const float*)d_in[11];
  float* out = (float*)d_out;

  const int E = in_sizes[0] / 11;

  // workspace: two bf16 activation buffers (ping-pong) + bf16 weights
  u16* curA = (u16*)d_ws;
  u16* curB = curA + (size_t)E * 64;
  u16* Wb   = curB + (size_t)E * 64;
  const int WTOT = 3 * 64 * 320;

  prep_wbf16<<<(WTOT + 255) / 256, 256, 0, stream>>>(W_rest, Wb, WTOT);

  const int gEdge = (E + 255) / 256;
  layer0_kernel<<<gEdge, 256, 0, stream>>>(x, nbr, W0, b0, gamma, beta, curA, E);

  const int gTile = (E + 127) / 128;
  // layer 1: A -> B
  mesh_layer_kernel<<<gTile, 256, 0, stream>>>(curA, curB, nbr, Wb + 0 * 20480,
                                               b_rest + 0 * 64, gamma + 1 * 64, beta + 1 * 64, E);
  // layer 2: B -> A
  mesh_layer_kernel<<<gTile, 256, 0, stream>>>(curB, curA, nbr, Wb + 1 * 20480,
                                               b_rest + 1 * 64, gamma + 2 * 64, beta + 2 * 64, E);
  // layer 3: A -> B
  mesh_layer_kernel<<<gTile, 256, 0, stream>>>(curA, curB, nbr, Wb + 2 * 20480,
                                               b_rest + 2 * 64, gamma + 3 * 64, beta + 3 * 64, E);

  classifier_kernel<<<gEdge, 256, 0, stream>>>(curB, Wc1, bc1, Wc2, bc2, out, E);
}
